// DiffusionTransformerMamba_36945308680370
// MI455X (gfx1250) — compile-verified, matrix operand order checked by reading, not run on hardware
//
#include <hip/hip_runtime.h>
#include <hip/hip_bf16.h>
#include <math.h>

// ---------------------------------------------------------------------------
// Problem constants (match reference)
// ---------------------------------------------------------------------------
#define BATCH   2
#define SEQ     1443
#define MTOK    (BATCH * SEQ)          // 2886 tokens
#define DMODEL  768
#define DINNER  1536
#define DSTATE  16
#define DCONV   4
#define DTRANK  48
#define XPROJN  (DTRANK + 2 * DSTATE)  // 80
#define NHEADS  8
#define HDIM    96
#define CONDL   256
#define KVTOK   (BATCH * CONDL)        // 512
#define NLAYER  4
#define EDGEF   9

// ---------------------------------------------------------------------------
// WMMA types (CDNA5 gfx1250, wave32)
// ---------------------------------------------------------------------------
typedef __attribute__((ext_vector_type(16))) __bf16 v16bf;
typedef __attribute__((ext_vector_type(8)))  __bf16 v8bf;
typedef __attribute__((ext_vector_type(8)))  float  v8f;
typedef __attribute__((ext_vector_type(4)))  float  v4f;
typedef __attribute__((ext_vector_type(4)))  unsigned int v4u;

union Frag { v16bf v; v8bf h[2]; };

#define BM 128
#define BN 128
#define BK 32
#define KPAD 48   // bf16 elems per LDS row (96B stride: 16B aligned, bank-staggered)

__device__ __forceinline__ float act_apply(float v, int act) {
  if (act == 1) return v / (1.f + expf(-v));                                // silu
  if (act == 2) return 0.5f * v * (1.f + erff(v * 0.70710678118654752f));   // gelu (exact)
  if (act == 3) return (v > 20.f) ? v : log1pf(expf(v));                    // softplus
  return v;
}

// pack two f32 into one dword of bf16 pairs (lowers to v_cvt_pk_bf16_f32)
__device__ __forceinline__ unsigned pack2(float a, float b) {
  union { __bf16 h[2]; unsigned u; } t;
  t.h[0] = (__bf16)a; t.h[1] = (__bf16)b;
  return t.u;
}

// pack 16 staged f32 (4 x v4f) -> 16 bf16 and store as 2 x ds_store_b128
__device__ __forceinline__ void store16(__bf16* dst, const v4f r[4]) {
  v4u u0, u1;
  u0.x = pack2(r[0].x, r[0].y); u0.y = pack2(r[0].z, r[0].w);
  u0.z = pack2(r[1].x, r[1].y); u0.w = pack2(r[1].z, r[1].w);
  u1.x = pack2(r[2].x, r[2].y); u1.y = pack2(r[2].z, r[2].w);
  u1.z = pack2(r[3].x, r[3].y); u1.w = pack2(r[3].z, r[3].w);
  *reinterpret_cast<v4u*>(dst)     = u0;
  *reinterpret_cast<v4u*>(dst + 8) = u1;
}

// ---------------------------------------------------------------------------
// General GEMM:  C[M,N] = act(A[M,K] @ W[N,K]^T + bias) + res
// Batched over grid.z (z -> (b, h) with per-b/per-h pointer strides) so the
// same kernel does token GEMMs, attention scores, and probs@V (btrans).
// f32 operands converted to bf16 while staging into double-buffered LDS;
// consumed by v_wmma_f32_16x16x32_bf16 (f32 accumulation). Software pipeline:
// issue global b128 loads for tile k+1, run WMMAs for tile k, then pack+store.
// ---------------------------------------------------------------------------
struct GemmArgs {
  const float* A;  long lda, sAb, sAh;
  const float* B;  long ldb, sBb, sBh;
  const float* bias;
  const float* res; long ldres, sRb, sRh;
  float* C; long ldc, sCb, sCh;
  int M, N, K, Z, H, act, btrans;
};

__global__ __launch_bounds__(256) void gemm_wmma_k(GemmArgs g) {
  __shared__ __attribute__((aligned(16))) __bf16 As[2][BM * KPAD];
  __shared__ __attribute__((aligned(16))) __bf16 Bs[2][BN * KPAD];

  const int z  = blockIdx.z;
  const int bb = z / g.H, hh = z % g.H;
  const float* Ap = g.A + (long)bb * g.sAb + (long)hh * g.sAh;
  const float* Bp = g.B + (long)bb * g.sBb + (long)hh * g.sBh;
  float*       Cp = g.C + (long)bb * g.sCb + (long)hh * g.sCh;
  const float* Rp = g.res ? g.res + (long)bb * g.sRb + (long)hh * g.sRh : nullptr;

  const int m0 = blockIdx.y * BM;
  const int n0 = blockIdx.x * BN;
  const int tid  = threadIdx.x;
  const int wave = tid >> 5, lane = tid & 31;
  const int wm = wave >> 1, wn = wave & 1;   // 4 x 2 wave grid -> 32x64 per wave
  const int lm = lane & 15, lk = lane >> 4;  // fragment row, K-half select

  const int srow = tid >> 1;          // staging row 0..127
  const int scol = (tid & 1) * 16;    // staging col 0 or 16

  const bool fullM = (m0 + BM <= g.M);
  const bool fullN = (n0 + BN <= g.N) && !g.btrans;

  v8f acc[2][4] = {};
  v4f ra[4], rb[4];
  bool fA = false, fB = false;

  const int ktiles = (g.K + BK - 1) / BK;

  auto load_tiles = [&](int kt) {
    const int k0 = kt * BK;
    const bool fullK = (k0 + BK <= g.K);
    fA = fullM && fullK;
    fB = fullN && fullK;
    if (fA) {
      const float* p = Ap + (long)(m0 + srow) * g.lda + k0 + scol;
      #pragma unroll
      for (int q = 0; q < 4; ++q) ra[q] = *reinterpret_cast<const v4f*>(p + 4 * q);
      if (k0 + 2 * BK <= g.K) __builtin_prefetch(p + 2 * BK, 0, 1);
    }
    if (fB) {
      const float* p = Bp + (long)(n0 + srow) * g.ldb + k0 + scol;
      #pragma unroll
      for (int q = 0; q < 4; ++q) rb[q] = *reinterpret_cast<const v4f*>(p + 4 * q);
      if (k0 + 2 * BK <= g.K) __builtin_prefetch(p + 2 * BK, 0, 1);
    }
  };

  auto store_tiles = [&](int buf, int kt) {
    const int k0 = kt * BK;
    if (fA) {
      store16(&As[buf][srow * KPAD + scol], ra);
    } else {
      for (int i = tid; i < BM * BK; i += 256) {
        const int r = i >> 5, c = i & 31;
        const int gm = m0 + r, gk = k0 + c;
        float v = 0.f;
        if (gm < g.M && gk < g.K) v = Ap[(long)gm * g.lda + gk];
        As[buf][r * KPAD + c] = (__bf16)v;
      }
    }
    if (fB) {
      store16(&Bs[buf][srow * KPAD + scol], rb);
    } else {
      for (int i = tid; i < BN * BK; i += 256) {
        const int r = i >> 5, c = i & 31;
        const int gn = n0 + r, gk = k0 + c;
        float v = 0.f;
        if (gn < g.N && gk < g.K)
          v = g.btrans ? Bp[(long)gk * g.ldb + gn] : Bp[(long)gn * g.ldb + gk];
        Bs[buf][r * KPAD + c] = (__bf16)v;
      }
    }
  };

  auto compute = [&](int buf) {
    // Fragment loads matching 16-bit WMMA operand layout:
    //   lanes 0-15: K = {0..7, 16..23}; lanes 16-31: K = {8..15, 24..31}
    Frag fa[2], fb[4];
    #pragma unroll
    for (int i = 0; i < 2; ++i) {
      const __bf16* p = &As[buf][(wm * 32 + i * 16 + lm) * KPAD + lk * 8];
      fa[i].h[0] = *reinterpret_cast<const v8bf*>(p);
      fa[i].h[1] = *reinterpret_cast<const v8bf*>(p + 16);
    }
    #pragma unroll
    for (int j = 0; j < 4; ++j) {
      const __bf16* p = &Bs[buf][(wn * 64 + j * 16 + lm) * KPAD + lk * 8];
      fb[j].h[0] = *reinterpret_cast<const v8bf*>(p);
      fb[j].h[1] = *reinterpret_cast<const v8bf*>(p + 16);
    }
    #pragma unroll
    for (int i = 0; i < 2; ++i)
      #pragma unroll
      for (int j = 0; j < 4; ++j)
        acc[i][j] = __builtin_amdgcn_wmma_f32_16x16x32_bf16(
            false, fa[i].v, false, fb[j].v, (short)0, acc[i][j], false, false);
  };

  // ---- software-pipelined main loop (double-buffered LDS) ----
  int cur = 0;
  load_tiles(0);
  store_tiles(cur, 0);
  __syncthreads();
  for (int kt = 0; kt < ktiles; ++kt) {
    const bool hasNext = (kt + 1 < ktiles);
    if (hasNext) load_tiles(kt + 1);    // global loads in flight during WMMAs
    compute(cur);
    if (hasNext) store_tiles(cur ^ 1, kt + 1);
    __syncthreads();
    cur ^= 1;
  }

  // Epilogue: C/D layout -> row = r + 8*lk, col = lm within each 16x16 tile
  #pragma unroll
  for (int i = 0; i < 2; ++i)
    for (int j = 0; j < 4; ++j)
      #pragma unroll
      for (int r = 0; r < 8; ++r) {
        const int gm = m0 + wm * 32 + i * 16 + r + 8 * lk;
        const int gn = n0 + wn * 64 + j * 16 + lm;
        if (gm < g.M && gn < g.N) {
          float v = acc[i][j][r];
          if (g.bias) v += g.bias[gn];
          v = act_apply(v, g.act);
          if (Rp) v += Rp[(long)gm * g.ldres + gn];
          Cp[(long)gm * g.ldc + gn] = v;
        }
      }
}

// ---------------------------------------------------------------------------
// LayerNorm: one wave32 per row, 8 rows per block
// ---------------------------------------------------------------------------
__global__ __launch_bounds__(256) void layernorm_k(const float* x, const float* w,
                                                   const float* b, float* y,
                                                   int rows, int cols) {
  const int row  = blockIdx.x * 8 + (threadIdx.x >> 5);
  const int lane = threadIdx.x & 31;
  if (row >= rows) return;
  const float* xr = x + (long)row * cols;
  float s = 0.f;
  for (int c = lane; c < cols; c += 32) s += xr[c];
  #pragma unroll
  for (int o = 16; o > 0; o >>= 1) s += __shfl_xor(s, o, 32);
  const float mu = s / cols;
  float v = 0.f;
  for (int c = lane; c < cols; c += 32) { float d = xr[c] - mu; v += d * d; }
  #pragma unroll
  for (int o = 16; o > 0; o >>= 1) v += __shfl_xor(v, o, 32);
  const float rs = rsqrtf(v / cols + 1e-5f);
  float* yr = y + (long)row * cols;
  for (int c = lane; c < cols; c += 32) yr[c] = (xr[c] - mu) * rs * w[c] + b[c];
}

// Softmax over 256 columns, wave per row, fused 1/sqrt(hd) scale
__global__ __launch_bounds__(256) void softmax_k(float* s, int rows, float scale) {
  const int row  = blockIdx.x * 8 + (threadIdx.x >> 5);
  const int lane = threadIdx.x & 31;
  if (row >= rows) return;
  float* sr = s + (long)row * CONDL;
  float v[8];
  float mx = -1e30f;
  #pragma unroll
  for (int i = 0; i < 8; ++i) { v[i] = sr[lane + 32 * i] * scale; mx = fmaxf(mx, v[i]); }
  #pragma unroll
  for (int o = 16; o > 0; o >>= 1) mx = fmaxf(mx, __shfl_xor(mx, o, 32));
  float sum = 0.f;
  #pragma unroll
  for (int i = 0; i < 8; ++i) { v[i] = expf(v[i] - mx); sum += v[i]; }
  #pragma unroll
  for (int o = 16; o > 0; o >>= 1) sum += __shfl_xor(sum, o, 32);
  const float inv = 1.f / sum;
  #pragma unroll
  for (int i = 0; i < 8; ++i) sr[lane + 32 * i] = v[i] * inv;
}

// Depthwise causal conv1d (d_conv=4) + bias + silu. xi = xz[:, :DINNER], lda=2*DINNER
__global__ __launch_bounds__(256) void conv_silu_k(const float* xz, const float* w,
                                                   const float* bias, float* xc) {
  const long idx = (long)blockIdx.x * 256 + threadIdx.x;
  if (idx >= (long)MTOK * DINNER) return;
  const int  d = (int)(idx % DINNER);
  const long m = idx / DINNER;
  const int  l = (int)(m % SEQ);
  float a = bias[d];
  #pragma unroll
  for (int j = 0; j < DCONV; ++j) {
    const int lj = l - (DCONV - 1) + j;
    if (lj >= 0) a += w[d * DCONV + j] * xz[(m - (DCONV - 1) + j) * (long)(2 * DINNER) + d];
  }
  xc[idx] = a / (1.f + expf(-a));
}

// y *= silu(z),  z = xz[:, DINNER:2*DINNER]
__global__ __launch_bounds__(256) void gate_silu_k(float* y, const float* xz) {
  const long idx = (long)blockIdx.x * 256 + threadIdx.x;
  if (idx >= (long)MTOK * DINNER) return;
  const long m = idx / DINNER;
  const int  d = (int)(idx % DINNER);
  const float z = xz[m * (long)(2 * DINNER) + DINNER + d];
  y[idx] *= z / (1.f + expf(-z));
}

// Selective scan: thread = one channel d; 16 states in registers; B/C via LDS.
// dA, dBx formed on the fly; y = sum_s h*C + xc*D
__global__ __launch_bounds__(256) void scan_k(const float* dt, const float* xc,
                                              const float* xdb, const float* A_log,
                                              const float* Dp, float* y) {
  __shared__ float sB[DSTATE], sC[DSTATE];
  const int d = blockIdx.x * 256 + threadIdx.x;   // DINNER = 6*256, no guard needed
  const int b = blockIdx.y;
  float a[DSTATE], h[DSTATE];
  #pragma unroll
  for (int s = 0; s < DSTATE; ++s) { a[s] = -expf(A_log[d * DSTATE + s]); h[s] = 0.f; }
  const float Dv = Dp[d];
  for (int l = 0; l < SEQ; ++l) {
    const long m = (long)b * SEQ + l;
    if (threadIdx.x < DSTATE)
      sB[threadIdx.x] = xdb[m * XPROJN + DTRANK + threadIdx.x];
    else if (threadIdx.x < 2 * DSTATE)
      sC[threadIdx.x - DSTATE] = xdb[m * XPROJN + DTRANK + DSTATE + (threadIdx.x - DSTATE)];
    __syncthreads();
    const float dtv = dt[m * (long)DINNER + d];
    const float xcv = xc[m * (long)DINNER + d];
    float yv = 0.f;
    #pragma unroll
    for (int s = 0; s < DSTATE; ++s) {
      const float dA = __expf(dtv * a[s]);
      h[s] = dA * h[s] + (dtv * sB[s]) * xcv;
      yv += h[s] * sC[s];
    }
    y[m * (long)DINNER + d] = yv + xcv * Dv;
    __syncthreads();
  }
}

// Flip along L (per batch), with independent src/dst leading dims (dst col offset
// folded into dst pointer)
__global__ __launch_bounds__(256) void flipL_k(const float* src, long lds_, float* dst,
                                               long ldd, int cols) {
  const long idx = (long)blockIdx.x * 256 + threadIdx.x;
  if (idx >= (long)MTOK * cols) return;
  const int  c = (int)(idx % cols);
  const long m = idx / cols;
  const int  b = (int)(m / SEQ);
  const int  l = (int)(m % SEQ);
  dst[((long)b * SEQ + (SEQ - 1 - l)) * ldd + c] = src[m * lds_ + c];
}

// x = x_in + emb[b] (broadcast over L)
__global__ __launch_bounds__(256) void add_emb_k(const float* xin, const float* emb, float* x) {
  const long idx = (long)blockIdx.x * 256 + threadIdx.x;
  if (idx >= (long)MTOK * DMODEL) return;
  const int c = (int)(idx % DMODEL);
  const int b = (int)((idx / DMODEL) / SEQ);
  x[idx] = xin[idx] + emb[b * DMODEL + c];
}

// sinusoidal timestep embedding: e[b, 0:384]=sin, e[b, 384:768]=cos
__global__ void ts_embed_k(const int* t, float* e) {
  const int b = blockIdx.x;
  const int j = threadIdx.x;        // 384 threads
  const int half = DMODEL / 2;
  const float freq = expf((float)j * (-logf(10000.f) / (float)(half - 1)));
  const float a = (float)t[b] * freq;
  e[b * DMODEL + j]        = sinf(a);
  e[b * DMODEL + half + j] = cosf(a);
}

// ---------------------------------------------------------------------------
// Host side
// ---------------------------------------------------------------------------
static void launch_gemm(hipStream_t st,
                        const float* A, long lda, const float* B, long ldb, int btrans,
                        const float* bias, const float* res, long ldres,
                        float* C, long ldc, int M, int N, int K, int act,
                        long sAb = 0, long sAh = 0, long sBb = 0, long sBh = 0,
                        long sRb = 0, long sRh = 0, long sCb = 0, long sCh = 0,
                        int Z = 1, int H = 1) {
  GemmArgs g;
  g.A = A; g.lda = lda; g.sAb = sAb; g.sAh = sAh;
  g.B = B; g.ldb = ldb; g.sBb = sBb; g.sBh = sBh;
  g.bias = bias;
  g.res = res; g.ldres = ldres; g.sRb = sRb; g.sRh = sRh;
  g.C = C; g.ldc = ldc; g.sCb = sCb; g.sCh = sCh;
  g.M = M; g.N = N; g.K = K; g.Z = Z; g.H = H; g.act = act; g.btrans = btrans;
  dim3 grid((N + BN - 1) / BN, (M + BM - 1) / BM, Z);
  gemm_wmma_k<<<grid, dim3(256), 0, st>>>(g);
}

static inline dim3 ew_grid(long n) { return dim3((unsigned)((n + 255) / 256)); }

// d_in flatten order (jax tree: dicts sorted by key):
// x, cond, timesteps, then params {final_norm_b, final_norm_w, head_b, head_w,
// layers{...sorted, with mb before mf...}, t_b1, t_b2, t_w1, t_w2}
enum {
  IN_X = 0, IN_COND, IN_TS,
  P_FINAL_NORM_B, P_FINAL_NORM_W, P_HEAD_B, P_HEAD_W,
  P_ATTN_IN_B, P_ATTN_IN_W, P_ATTN_OUT_B, P_ATTN_OUT_W,
  P_CROSS_NORM_B, P_CROSS_NORM_W,
  P_FFN2_B1, P_FFN2_B2, P_FFN2_NORM_B, P_FFN2_NORM_W, P_FFN2_W1, P_FFN2_W2,
  P_FFN_B1, P_FFN_B2, P_FFN_NORM_B, P_FFN_NORM_W, P_FFN_W1, P_FFN_W2,
  MB_A_LOG, MB_D, MB_CONV_B, MB_CONV_W, MB_DT_B, MB_DT_W, MB_IN_W, MB_OUT_W, MB_X_W,
  MF_A_LOG, MF_D, MF_CONV_B, MF_CONV_W, MF_DT_B, MF_DT_W, MF_IN_W, MF_OUT_W, MF_X_W,
  P_MIX_B, P_MIX_W, P_NORM_B, P_NORM_W,
  P_T_B1, P_T_B2, P_T_W1, P_T_W2
};

extern "C" void kernel_launch(void* const* d_in, const int* in_sizes, int n_in,
                              void* d_out, int out_size, void* d_ws, size_t ws_size,
                              hipStream_t stream) {
  (void)in_sizes; (void)n_in; (void)out_size; (void)ws_size;
  const float* x_in = (const float*)d_in[IN_X];
  const float* cond = (const float*)d_in[IN_COND];
  const int*   tst  = (const int*)d_in[IN_TS];
  float* out = (float*)d_out;

  auto P = [&](int i) { return (const float*)d_in[i]; };

  // ---- workspace layout (floats) ----
  float* ws = (float*)d_ws;
  long off = 0;
  auto alloc = [&](long n) { float* p = ws + off; off += n; return p; };
  float* EMB_E = alloc((long)BATCH * DMODEL);
  float* EMB_H = alloc((long)BATCH * 4 * DMODEL);
  float* EMB   = alloc((long)BATCH * DMODEL);
  float* X     = alloc((long)MTOK * DMODEL);
  float* XN    = alloc((long)MTOK * DMODEL);
  float* XNF   = alloc((long)MTOK * DMODEL);
  float* XZ    = alloc((long)MTOK * 2 * DINNER);   // also reused as FFN hidden [M,3072]
  float* XC    = alloc((long)MTOK * DINNER);
  float* XDB   = alloc((long)MTOK * XPROJN);
  float* DTB   = alloc((long)MTOK * DINNER);
  float* Y     = alloc((long)MTOK * DINNER);
  float* CAT   = alloc((long)MTOK * DINNER);
  float* B768  = alloc((long)MTOK * DMODEL);
  float* ATT   = alloc((long)BATCH * NHEADS * SEQ * CONDL);
  float* Q     = alloc((long)MTOK * DMODEL);
  float* KB    = alloc((long)KVTOK * DMODEL);
  float* VB    = alloc((long)KVTOK * DMODEL);
  float* O     = alloc((long)MTOK * DMODEL);

  // ---- timestep embedding: e -> silu(e W1^T + b1) -> W2^T + b2, broadcast add ----
  ts_embed_k<<<BATCH, DMODEL / 2, 0, stream>>>(tst, EMB_E);
  launch_gemm(stream, EMB_E, DMODEL, P(P_T_W1), DMODEL, 0, P(P_T_B1), nullptr, 0,
              EMB_H, 4 * DMODEL, BATCH, 4 * DMODEL, DMODEL, /*silu*/1);
  launch_gemm(stream, EMB_H, 4 * DMODEL, P(P_T_W2), 4 * DMODEL, 0, P(P_T_B2), nullptr, 0,
              EMB, DMODEL, BATCH, DMODEL, 4 * DMODEL, 0);
  add_emb_k<<<ew_grid((long)MTOK * DMODEL), 256, 0, stream>>>(x_in, EMB, X);

  // one mamba direction (input tokens -> out[M, DMODEL] with given ldc)
  auto run_mamba = [&](const float* input, int base, int l, float* dst, long dstLdc) {
    const float* A_log = P(base + 0) + (long)l * DINNER * DSTATE;
    const float* Dp    = P(base + 1) + (long)l * DINNER;
    const float* convB = P(base + 2) + (long)l * DINNER;
    const float* convW = P(base + 3) + (long)l * DINNER * DCONV;
    const float* dtB   = P(base + 4) + (long)l * DINNER;
    const float* dtW   = P(base + 5) + (long)l * DINNER * DTRANK;
    const float* inW   = P(base + 6) + (long)l * 2 * DINNER * DMODEL;
    const float* outW  = P(base + 7) + (long)l * DMODEL * DINNER;
    const float* xW    = P(base + 8) + (long)l * XPROJN * DINNER;

    launch_gemm(stream, input, DMODEL, inW, DMODEL, 0, nullptr, nullptr, 0,
                XZ, 2 * DINNER, MTOK, 2 * DINNER, DMODEL, 0);
    conv_silu_k<<<ew_grid((long)MTOK * DINNER), 256, 0, stream>>>(XZ, convW, convB, XC);
    launch_gemm(stream, XC, DINNER, xW, DINNER, 0, nullptr, nullptr, 0,
                XDB, XPROJN, MTOK, XPROJN, DINNER, 0);
    launch_gemm(stream, XDB, XPROJN, dtW, DTRANK, 0, dtB, nullptr, 0,
                DTB, DINNER, MTOK, DINNER, DTRANK, /*softplus*/3);
    scan_k<<<dim3(DINNER / 256, BATCH), 256, 0, stream>>>(DTB, XC, XDB, A_log, Dp, Y);
    gate_silu_k<<<ew_grid((long)MTOK * DINNER), 256, 0, stream>>>(Y, XZ);
    launch_gemm(stream, Y, DINNER, outW, DINNER, 0, nullptr, nullptr, 0,
                dst, dstLdc, MTOK, DMODEL, DINNER, 0);
  };

  const int lnGrid = (MTOK + 7) / 8;

  for (int l = 0; l < NLAYER; ++l) {
    // ---- bidirectional mamba block ----
    layernorm_k<<<lnGrid, 256, 0, stream>>>(X, P(P_NORM_W) + l * DMODEL,
                                            P(P_NORM_B) + l * DMODEL, XN, MTOK, DMODEL);
    flipL_k<<<ew_grid((long)MTOK * DMODEL), 256, 0, stream>>>(XN, DMODEL, XNF, DMODEL, DMODEL);
    run_mamba(XN,  MF_A_LOG, l, CAT, DINNER);            // forward -> CAT[:, 0:768]
    run_mamba(XNF, MB_A_LOG, l, B768, DMODEL);           // backward in flipped space
    flipL_k<<<ew_grid((long)MTOK * DMODEL), 256, 0, stream>>>(B768, DMODEL,
                                                              CAT + DMODEL, DINNER, DMODEL);
    launch_gemm(stream, CAT, DINNER, P(P_MIX_W) + (long)l * DMODEL * DINNER, DINNER, 0,
                P(P_MIX_B) + l * DMODEL, X, DMODEL, X, DMODEL, MTOK, DMODEL, DINNER, 0);

    // ---- FFN 1 ----
    layernorm_k<<<lnGrid, 256, 0, stream>>>(X, P(P_FFN_NORM_W) + l * DMODEL,
                                            P(P_FFN_NORM_B) + l * DMODEL, XN, MTOK, DMODEL);
    launch_gemm(stream, XN, DMODEL, P(P_FFN_W1) + (long)l * 4 * DMODEL * DMODEL, DMODEL, 0,
                P(P_FFN_B1) + l * 4 * DMODEL, nullptr, 0,
                XZ, 4 * DMODEL, MTOK, 4 * DMODEL, DMODEL, /*gelu*/2);
    launch_gemm(stream, XZ, 4 * DMODEL, P(P_FFN_W2) + (long)l * DMODEL * 4 * DMODEL,
                4 * DMODEL, 0, P(P_FFN_B2) + l * DMODEL, X, DMODEL,
                X, DMODEL, MTOK, DMODEL, 4 * DMODEL, 0);

    // ---- cross attention ----
    layernorm_k<<<lnGrid, 256, 0, stream>>>(X, P(P_CROSS_NORM_W) + l * DMODEL,
                                            P(P_CROSS_NORM_B) + l * DMODEL, XN, MTOK, DMODEL);
    const float* wIn = P(P_ATTN_IN_W) + (long)l * 3 * DMODEL * DMODEL;
    const float* bIn = P(P_ATTN_IN_B) + (long)l * 3 * DMODEL;
    launch_gemm(stream, XN, DMODEL, wIn, DMODEL, 0, bIn, nullptr, 0,
                Q, DMODEL, MTOK, DMODEL, DMODEL, 0);
    launch_gemm(stream, cond, DMODEL, wIn + (long)DMODEL * DMODEL, DMODEL, 0,
                bIn + DMODEL, nullptr, 0, KB, DMODEL, KVTOK, DMODEL, DMODEL, 0);
    launch_gemm(stream, cond, DMODEL, wIn + (long)2 * DMODEL * DMODEL, DMODEL, 0,
                bIn + 2 * DMODEL, nullptr, 0, VB, DMODEL, KVTOK, DMODEL, DMODEL, 0);
    // scores[b,h] = Q_slice @ K_slice^T   (M=1443, N=256, K=96)
    launch_gemm(stream, Q, DMODEL, KB, DMODEL, 0, nullptr, nullptr, 0,
                ATT, CONDL, SEQ, CONDL, HDIM, 0,
                /*sAb*/(long)SEQ * DMODEL, /*sAh*/HDIM,
                /*sBb*/(long)CONDL * DMODEL, /*sBh*/HDIM,
                0, 0,
                /*sCb*/(long)NHEADS * SEQ * CONDL, /*sCh*/(long)SEQ * CONDL,
                BATCH * NHEADS, NHEADS);
    softmax_k<<<(BATCH * NHEADS * SEQ + 7) / 8, 256, 0, stream>>>(
        ATT, BATCH * NHEADS * SEQ, rsqrtf((float)HDIM));
    // o[b,h] = probs @ V_slice  (btrans: W[n,k] = V[k*ldb + n])
    launch_gemm(stream, ATT, CONDL, VB, DMODEL, 1, nullptr, nullptr, 0,
                O, DMODEL, SEQ, HDIM, CONDL, 0,
                /*sAb*/(long)NHEADS * SEQ * CONDL, /*sAh*/(long)SEQ * CONDL,
                /*sBb*/(long)CONDL * DMODEL, /*sBh*/HDIM,
                0, 0,
                /*sCb*/(long)SEQ * DMODEL, /*sCh*/HDIM,
                BATCH * NHEADS, NHEADS);
    launch_gemm(stream, O, DMODEL, P(P_ATTN_OUT_W) + (long)l * DMODEL * DMODEL, DMODEL, 0,
                P(P_ATTN_OUT_B) + l * DMODEL, X, DMODEL, X, DMODEL, MTOK, DMODEL, DMODEL, 0);

    // ---- FFN 2 ----
    layernorm_k<<<lnGrid, 256, 0, stream>>>(X, P(P_FFN2_NORM_W) + l * DMODEL,
                                            P(P_FFN2_NORM_B) + l * DMODEL, XN, MTOK, DMODEL);
    launch_gemm(stream, XN, DMODEL, P(P_FFN2_W1) + (long)l * 4 * DMODEL * DMODEL, DMODEL, 0,
                P(P_FFN2_B1) + l * 4 * DMODEL, nullptr, 0,
                XZ, 4 * DMODEL, MTOK, 4 * DMODEL, DMODEL, /*gelu*/2);
    launch_gemm(stream, XZ, 4 * DMODEL, P(P_FFN2_W2) + (long)l * DMODEL * 4 * DMODEL,
                4 * DMODEL, 0, P(P_FFN2_B2) + l * DMODEL, X, DMODEL,
                X, DMODEL, MTOK, DMODEL, 4 * DMODEL, 0);
  }

  // ---- final norm + head ----
  layernorm_k<<<lnGrid, 256, 0, stream>>>(X, P(P_FINAL_NORM_W), P(P_FINAL_NORM_B),
                                          XN, MTOK, DMODEL);
  launch_gemm(stream, XN, DMODEL, P(P_HEAD_W), DMODEL, 0, P(P_HEAD_B), nullptr, 0,
              out, EDGEF, MTOK, EDGEF, DMODEL, 0);
}
